// PerlinAttention_81432579932369
// MI455X (gfx1250) — matrix operand order repaired and checked by reading, not compiled, not run on hardware
//
#include <hip/hip_runtime.h>
#include <hip/hip_bf16.h>
#include <math.h>

typedef __attribute__((ext_vector_type(16))) _Float16 v16h;
typedef __attribute__((ext_vector_type(8)))  _Float16 v8h;
typedef __attribute__((ext_vector_type(8)))  float    v8f;

namespace {

constexpr int Nb = 4, Hn = 12, Tt = 1024, Hd = 64;
constexpr int Mr = 266, Mp = 288;                 // FAVOR features, padded to 9*32
constexpr int Kk = 128, Ee = 128, Pvd = 192;
constexpr int NHb = Nb * Hn;                      // 48 batches
constexpr int Rows = NHb * Tt;                    // 49152 token rows
__device__ constexpr float FP_MIN = -1.7014118e38f;

__device__ __forceinline__ v16h cat8(v8h lo, v8h hi) {
  return __builtin_shufflevector(lo, hi, 0, 1, 2, 3, 4, 5, 6, 7,
                                 8, 9, 10, 11, 12, 13, 14, 15);
}

// =====================================================================
// Batched WMMA GEMM, both operands K-major:
//   C[M x Ncols] = A[M x K] * BT[Ncols x K]^T   (A,BT f16; C f32)
// Fragment loads are 2x b128 per operand per lane (contiguous k order, same for A and B):
//   A lane(m,half):  A[m][k0+16h .. +16)
//   B lane(n,half):  BT[n][k0+16h .. +16)
// grid: (Ncols/32, ceil(Mtiles/4), batch); block 128 = 4 waves;
// each wave computes a 16x32 tile (one A frag feeds two WMMAs).
// =====================================================================
__global__ void wmma_gemm_kernel(const _Float16* __restrict__ A,
                                 const _Float16* __restrict__ BT,
                                 float* __restrict__ C,
                                 int Mrows, int Kdim,
                                 long strideA, long strideB, long strideC,
                                 int lda, int ldb, int ldc)
{
  const int lane = threadIdx.x & 31;
  const int wave = threadIdx.x >> 5;
  const int batch = blockIdx.z;
  const int mt = blockIdx.y * 4 + wave;            // 16-row tile index
  if (mt * 16 >= Mrows) return;                    // wave-uniform exit
  const int mrow = lane & 15;
  const int half = lane >> 4;
  const int gm  = mt * 16 + mrow;
  const int gn0 = blockIdx.x * 32 + mrow;
  const int gn1 = gn0 + 16;

  const _Float16* __restrict__ Ab = A  + (long)batch * strideA;
  const _Float16* __restrict__ Bb = BT + (long)batch * strideB;
  float* __restrict__ Cb = C + (long)batch * strideC;

  const v8h* __restrict__ pa  = (const v8h*)(Ab + (long)gm  * lda) + 2 * half;
  const v8h* __restrict__ pb0 = (const v8h*)(Bb + (long)gn0 * ldb) + 2 * half;
  const v8h* __restrict__ pb1 = (const v8h*)(Bb + (long)gn1 * ldb) + 2 * half;

  v8f acc0 = {}, acc1 = {};
  for (int k0 = 0; k0 < Kdim; k0 += 32) {
    __builtin_prefetch((const void*)(pa + 4), 0, 1);    // speculative, next k-block
    v16h a  = cat8(pa[0],  pa[1]);
    v16h b0 = cat8(pb0[0], pb0[1]);
    v16h b1 = cat8(pb1[0], pb1[1]);
    acc0 = __builtin_amdgcn_wmma_f32_16x16x32_f16(false, a, false, b0,
                                                  (short)0, acc0, false, false);
    acc1 = __builtin_amdgcn_wmma_f32_16x16x32_f16(false, a, false, b1,
                                                  (short)0, acc1, false, false);
    pa += 4; pb0 += 4; pb1 += 4;
  }
#pragma unroll
  for (int i = 0; i < 8; ++i) {
    long rowoff = (long)(mt * 16 + i + 8 * half) * ldc;
    Cb[rowoff + gn0] = acc0[i];
    Cb[rowoff + gn1] = acc1[i];
  }
}

// ---------------- mask scan: zo cumsum, normalized position, pad flags -------
__global__ void scan_mask_kernel(const float* __restrict__ mask,
                                 float* __restrict__ tyn, float* __restrict__ padf)
{
  __shared__ float s[Tt];
  const int n = blockIdx.x, t = threadIdx.x;
  float m = mask[n * Tt + t];
  float zo = (m > -1.f) ? 1.f : 0.f;
  padf[n * Tt + t] = (m < -1.f) ? 1.f : 0.f;
  s[t] = zo;
  __syncthreads();
  for (int off = 1; off < Tt; off <<= 1) {
    float v = s[t];
    float add = (t >= off) ? s[t - off] : 0.f;
    __syncthreads();
    s[t] = v + add;
    __syncthreads();
  }
  float cum = s[t];
  float total = s[Tt - 1];
  tyn[n * Tt + t] = (cum - 1.f) / ((total - 1.f) + 1e-8f);
}

// --- build vaT (f16 [NH,128,T]: band-identity || v_for_atten, T-contig) and
//     vT (f16 [NH,64,T]: masked v, T-contig, B operand of final P@V) ----------
__global__ void build_va_kernel(const float* __restrict__ tyn,
                                const float* __restrict__ padf,
                                const float* __restrict__ v_att,
                                const float* __restrict__ v,
                                _Float16* __restrict__ vaT,
                                _Float16* __restrict__ vT)
{
  int idx = blockIdx.x * blockDim.x + threadIdx.x;       // over NH*T*128
  if (idx >= Rows * 128) return;
  int j = idx & 127;
  int row = idx >> 7;                                    // nh*T + t
  int t = row & (Tt - 1);
  int nh = row >> 10;
  int n = nh / Hn;
  float pad = padf[n * Tt + t];
  float val;
  if (j < Hd) {
    float gy = tyn[n * Tt + t] * (float)(Hd - 1);        // grid-sample of identity
    float y0f = floorf(gy);
    float wy = gy - y0f;
    int y0 = (int)y0f;
    val = 0.f;
    if (j == y0)     val += 1.f - wy;
    if (j == y0 + 1) val += wy;
    float vv = v[(long)row * Hd + j];
    vT[((long)nh * Hd + j) * Tt + t] = (_Float16)((pad > 0.5f) ? 0.f : vv);
  } else {
    val = v_att[(long)row * Hd + (j - Hd)];
  }
  if (pad > 0.5f) val = 0.f;
  vaT[((long)nh * 128 + j) * Tt + t] = (_Float16)val;
}

// ---------------- f32 -> f16 (optional scale) ---------------------------------
__global__ void cvt_scale16_kernel(const float* __restrict__ src,
                                   _Float16* __restrict__ dst, float scale, int total)
{
  int i = blockIdx.x * blockDim.x + threadIdx.x;
  if (i < total) dst[i] = (_Float16)(src[i] * scale);
}
__global__ void cvt16_kernel(const float* __restrict__ src,
                             _Float16* __restrict__ dst, int total)
{
  int i = blockIdx.x * blockDim.x + threadIdx.x;
  if (i < total) dst[i] = (_Float16)src[i];
}

// ------------- f32 -> f16 with zero row padding: out[Rp x C], src[R x C] ------
__global__ void cvt_pad_kernel(const float* __restrict__ src,
                               _Float16* __restrict__ dst, int R, int C, int Rp)
{
  int i = blockIdx.x * blockDim.x + threadIdx.x;
  if (i >= Rp * C) return;
  int r = i / C;
  dst[i] = (_Float16)((r < R) ? src[i] : 0.f);
}

// ---------------- diag = 0.5*sum(xn^2) = sum(x^2)/16 ---------------------------
__global__ void diag_kernel(const float* __restrict__ x, float* __restrict__ dg)
{
  int r = blockIdx.x * blockDim.x + threadIdx.x;
  if (r >= Rows) return;
  const float* p = x + (long)r * Hd;
  float s = 0.f;
  for (int i = 0; i < Hd; ++i) { float v = p[i]; s += v * v; }
  dg[r] = s * (1.f / 16.f);
}

// ---------------- per-row max over first Mr features --------------------------
__global__ void rowmax_kernel(const float* __restrict__ xp, float* __restrict__ rmax)
{
  __shared__ float red[128];
  int row = blockIdx.x, tid = threadIdx.x;
  const float* p = xp + (long)row * Mp;
  float mx = -3.4e38f;
  for (int m = tid; m < Mr; m += 128) mx = fmaxf(mx, p[m]);
  red[tid] = mx; __syncthreads();
  for (int off = 64; off > 0; off >>= 1) {
    if (tid < off) red[tid] = fmaxf(red[tid], red[tid + off]);
    __syncthreads();
  }
  if (tid == 0) rmax[row] = red[0];
}

__global__ void nhmax_kernel(const float* __restrict__ rmax, float* __restrict__ smax)
{
  __shared__ float red[256];
  int nh = blockIdx.x, tid = threadIdx.x;
  float mx = -3.4e38f;
  for (int t = tid; t < Tt; t += 256) mx = fmaxf(mx, rmax[nh * Tt + t]);
  red[tid] = mx; __syncthreads();
  for (int off = 128; off > 0; off >>= 1) {
    if (tid < off) red[tid] = fmaxf(red[tid], red[tid + off]);
    __syncthreads();
  }
  if (tid == 0) smax[nh] = red[0];
}

// ------- (exp(xp - diag - stab) + eps) * M^-0.5 -> f16; pad cols = 0 ----------
// transposed==0: out[row*Mp + m]  (qf, row-major A operand)
// transposed==1: out[(nh*Mp + m)*T + t]  (kfT, K-major B operand)
__global__ void favor_exp_kernel(const float* __restrict__ xp,
                                 const float* __restrict__ dg,
                                 const float* __restrict__ rmax,
                                 const float* __restrict__ smax,
                                 int use_smax, int transposed,
                                 _Float16* __restrict__ out)
{
  int idx = blockIdx.x * blockDim.x + threadIdx.x;       // rows * Mp
  if (idx >= Rows * Mp) return;
  int m = idx % Mp;
  int row = idx / Mp;
  float val = 0.f;
  if (m < Mr) {
    float stab = use_smax ? smax[row >> 10] : rmax[row];
    val = (expf(xp[idx] - dg[row] - stab) + 1e-4f) * 0.061313896f;  // 266^-0.5
  }
  if (transposed) {
    int t = row & (Tt - 1), nh = row >> 10;
    out[((long)nh * Mp + m) * Tt + t] = (_Float16)val;
  } else {
    out[(long)row * Mp + m] = (_Float16)val;
  }
}

// ---------------- kfsum[nh,m] = sum_t kfT[nh,m,t] (contiguous) ----------------
__global__ void kfsum_kernel(const _Float16* __restrict__ kfT, float* __restrict__ ks)
{
  int idx = blockIdx.x * blockDim.x + threadIdx.x;
  if (idx >= NHb * Mp) return;
  const _Float16* p = kfT + (long)idx * Tt;
  float s = 0.f;
  for (int t = 0; t < Tt; ++t) s += (float)p[t];
  ks[idx] = s;
}

// ---------------- z = 1 / (qf . kfsum) ----------------------------------------
__global__ void zdot_kernel(const _Float16* __restrict__ qf,
                            const float* __restrict__ ks, float* __restrict__ z)
{
  int row = blockIdx.x * blockDim.x + threadIdx.x;
  if (row >= Rows) return;
  int nh = row >> 10;
  float s = 0.f;
  for (int m = 0; m < Mp; ++m) s += (float)qf[(long)row * Mp + m] * ks[nh * Mp + m];
  z[row] = 1.f / s;
}

// ---------------- pv = [pcl*z || v] (f16, [NH,T,192] row-major A operand) -----
__global__ void build_pv_kernel(const float* __restrict__ pcl,
                                const float* __restrict__ z,
                                const _Float16* __restrict__ vT,
                                _Float16* __restrict__ pv16)
{
  int idx = blockIdx.x * blockDim.x + threadIdx.x;       // rows * 192
  if (idx >= Rows * Pvd) return;
  int e = idx % Pvd;
  int row = idx / Pvd;
  _Float16 val;
  if (e < 2 * Hd) {
    val = (_Float16)(pcl[(long)row * 2 * Hd + e] * z[row]);
  } else {
    int t = row & (Tt - 1), nh = row >> 10;
    val = vT[((long)nh * Hd + (e - 2 * Hd)) * Tt + t];
  }
  pv16[idx] = val;
}

// ---------------- layernorm(128) + exact gelu -> f16 --------------------------
__global__ void ln_gelu_kernel(const float* __restrict__ x,
                               const float* __restrict__ encb,
                               const float* __restrict__ g,
                               const float* __restrict__ b,
                               _Float16* __restrict__ out)
{
  __shared__ float red[128];
  int row = blockIdx.x, tid = threadIdx.x;
  float v = x[(long)row * Ee + tid] + encb[tid];
  red[tid] = v; __syncthreads();
  for (int off = 64; off > 0; off >>= 1) {
    if (tid < off) red[tid] += red[tid + off];
    __syncthreads();
  }
  float mu = red[0] * (1.f / 128.f);
  __syncthreads();
  float d = v - mu;
  red[tid] = d * d; __syncthreads();
  for (int off = 64; off > 0; off >>= 1) {
    if (tid < off) red[tid] += red[tid + off];
    __syncthreads();
  }
  float var = red[0] * (1.f / 128.f);
  float y = d * rsqrtf(var + 1e-5f) * g[tid] + b[tid];
  float ge = 0.5f * y * (1.f + erff(y * 0.70710678f));
  out[(long)row * Ee + tid] = (_Float16)ge;
}

// ---------------- x[..,k] += bias[k] ------------------------------------------
__global__ void add_rowbias_kernel(float* __restrict__ x, const float* __restrict__ bias,
                                   int ldk, int total)
{
  int i = blockIdx.x * blockDim.x + threadIdx.x;
  if (i < total) x[i] += bias[i % ldk];
}

// ---------------- direct 3x3 conv, pad 1; mode 0:none 1:relu 2:relu(skip+y) ---
__global__ void conv3x3_kernel(const float* __restrict__ x, const float* __restrict__ w,
                               const float* __restrict__ bias, const float* __restrict__ skip,
                               float* __restrict__ y,
                               int Bn, int Ci, int Hi, int Wi, int Co, int Ho, int Wo,
                               int stride, int mode)
{
  int idx = blockIdx.x * blockDim.x + threadIdx.x;
  int total = Bn * Co * Ho * Wo;
  if (idx >= total) return;
  int wo = idx % Wo; int t = idx / Wo;
  int ho = t % Ho; t /= Ho;
  int co = t % Co; int bb = t / Co;
  float acc = bias[co];
  for (int ci = 0; ci < Ci; ++ci) {
    const float* xp = x + ((long)(bb * Ci + ci) * Hi) * Wi;
    const float* wp = w + (long)(co * Ci + ci) * 9;
    for (int kh = 0; kh < 3; ++kh) {
      int ih = ho * stride - 1 + kh;
      if (ih < 0 || ih >= Hi) continue;
      for (int kw = 0; kw < 3; ++kw) {
        int iw = wo * stride - 1 + kw;
        if (iw < 0 || iw >= Wi) continue;
        acc += xp[ih * Wi + iw] * wp[kh * 3 + kw];
      }
    }
  }
  if (mode == 2) acc += skip[idx];
  if (mode != 0) acc = fmaxf(acc, 0.f);
  y[idx] = acc;
}

// ------ 3x3 conv over x2 nearest-upsampled input (fused): out [Bn,Co,2Hi,2Wi] --
__global__ void conv3x3_up_kernel(const float* __restrict__ x, const float* __restrict__ w,
                                  const float* __restrict__ bias, float* __restrict__ y,
                                  int Bn, int Ci, int Hi, int Wi, int Co)
{
  const int Ho = 2 * Hi, Wo = 2 * Wi;
  int idx = blockIdx.x * blockDim.x + threadIdx.x;
  int total = Bn * Co * Ho * Wo;
  if (idx >= total) return;
  int wo = idx % Wo; int t = idx / Wo;
  int ho = t % Ho; t /= Ho;
  int co = t % Co; int bb = t / Co;
  float acc = bias[co];
  for (int ci = 0; ci < Ci; ++ci) {
    const float* xp = x + ((long)(bb * Ci + ci) * Hi) * Wi;
    const float* wp = w + (long)(co * Ci + ci) * 9;
    for (int kh = 0; kh < 3; ++kh) {
      int ih = ho - 1 + kh;
      if (ih < 0 || ih >= Ho) continue;
      for (int kw = 0; kw < 3; ++kw) {
        int iw = wo - 1 + kw;
        if (iw < 0 || iw >= Wo) continue;
        acc += xp[(ih >> 1) * Wi + (iw >> 1)] * wp[kh * 3 + kw];
      }
    }
  }
  y[idx] = acc;
}

// ===== fused: linear-resize K=128->1024, mask, softmax, P(f16) @ V(f16) WMMA ==
// grid (T/16, H, N), block 128 (4 waves = 4 column tiles of 16)
__global__ void attn_out_kernel(const float* __restrict__ score,
                                const float* __restrict__ mask,
                                const _Float16* __restrict__ vT,
                                float* __restrict__ out)
{
  __shared__ float    srow[16][Kk];        // decoded score rows
  __shared__ float    red[16][8];
  __shared__ float    invsum[16];
  __shared__ _Float16 p16[16][Tt];         // unnormalized exp(probs) f16, K-major
  const int n = blockIdx.z, h = blockIdx.y, t0 = blockIdx.x * 16;
  const int tid = threadIdx.x;
  const long base = (((long)(n * Hn + h) * Tt) + t0) * Kk;
  for (int i = tid; i < 16 * Kk; i += 128)
    srow[i >> 7][i & (Kk - 1)] = score[base + (long)(i >> 7) * Kk + (i & (Kk - 1))];
  __syncthreads();

  const int r = tid >> 3, sub = tid & 7;
  const float* __restrict__ mrow = mask + (long)n * Tt;

  // pass 1: row max over resized+masked logits
  float mx = -3.4e38f;
  for (int s = sub * 128; s < sub * 128 + 128; ++s) {
    float src = (float)s * 0.125f - 0.4375f;   // half-pixel, scale 1/8
    float x0f = floorf(src);
    float wgt = src - x0f;
    int x0 = (int)x0f;
    int x0c = min(max(x0, 0), Kk - 1);
    int x1c = min(max(x0 + 1, 0), Kk - 1);
    float val = (1.f - wgt) * srow[r][x0c] + wgt * srow[r][x1c];
    if (mrow[s] < -1.f) val = FP_MIN;
    mx = fmaxf(mx, val);
  }
  red[r][sub] = mx; __syncthreads();
  float rowmax = red[r][0];
  for (int i = 1; i < 8; ++i) rowmax = fmaxf(rowmax, red[r][i]);
  __syncthreads();

  // pass 2: exp + sum, write f16 probabilities (unnormalized)
  float sm = 0.f;
  for (int s = sub * 128; s < sub * 128 + 128; ++s) {
    float src = (float)s * 0.125f - 0.4375f;
    float x0f = floorf(src);
    float wgt = src - x0f;
    int x0 = (int)x0f;
    int x0c = min(max(x0, 0), Kk - 1);
    int x1c = min(max(x0 + 1, 0), Kk - 1);
    float val = (1.f - wgt) * srow[r][x0c] + wgt * srow[r][x1c];
    if (mrow[s] < -1.f) val = FP_MIN;
    float e = expf(val - rowmax);
    sm += e;
    p16[r][s] = (_Float16)e;
  }
  red[r][sub] = sm; __syncthreads();
  if (sub == 0) {
    float s = 0.f;
    for (int i = 0; i < 8; ++i) s += red[r][i];
    invsum[r] = 1.f / s;
  }
  __syncthreads();

  // WMMA: ctx[16 x 64] = P[16 x 1024] @ V[1024 x 64]; V is K-major (vT)
  const int lane = tid & 31, wave = tid >> 5;
  const int mrw = lane & 15, half = lane >> 4;
  const v8h* __restrict__ pp = (const v8h*)(&p16[mrw][0]) + 2 * half;
  const v8h* __restrict__ pv =
      (const v8h*)(vT + ((long)(n * Hn + h) * Hd + wave * 16 + mrw) * Tt) + 2 * half;
  v8f acc = {};
  for (int k0 = 0; k0 < Tt; k0 += 32) {
    v16h a = cat8(pp[0], pp[1]);
    v16h b = cat8(pv[0], pv[1]);
    acc = __builtin_amdgcn_wmma_f32_16x16x32_f16(false, a, false, b,
                                                 (short)0, acc, false, false);
    pp += 4; pv += 4;
  }
#pragma unroll
  for (int i = 0; i < 8; ++i) {
    int row = i + 8 * half;
    out[((long)n * Tt + t0 + row) * (Hn * Hd) + h * Hd + wave * 16 + mrw] =
        acc[i] * invsum[row];
  }
}

inline int cdiv(int a, int b) { return (a + b - 1) / b; }

void launch_gemm(hipStream_t s, const _Float16* A, const _Float16* BT, float* C,
                 int M, int Ncols, int K, long sA, long sB, long sC,
                 int lda, int ldb, int ldc, int batch)
{
  dim3 grid(Ncols / 32, cdiv(M / 16, 4), batch);
  wmma_gemm_kernel<<<grid, 128, 0, s>>>(A, BT, C, M, K, sA, sB, sC, lda, ldb, ldc);
}

} // anonymous namespace

extern "C" void kernel_launch(void* const* d_in, const int* in_sizes, int n_in,
                              void* d_out, int out_size, void* d_ws, size_t ws_size,
                              hipStream_t stream) {
  (void)in_sizes; (void)n_in; (void)out_size; (void)ws_size;
  const float* v      = (const float*)d_in[2];
  const float* q_att  = (const float*)d_in[3];
  const float* k_att  = (const float*)d_in[4];
  const float* v_att  = (const float*)d_in[5];
  const float* mask   = (const float*)d_in[8];
  const float* proj   = (const float*)d_in[11];
  const float* enc_w  = (const float*)d_in[12];
  const float* enc_b  = (const float*)d_in[13];
  const float* ln_g   = (const float*)d_in[14];
  const float* ln_b   = (const float*)d_in[15];
  const float* dec_w  = (const float*)d_in[16];
  const float* dec_b  = (const float*)d_in[17];
  const float* c1_w   = (const float*)d_in[18];
  const float* c1_b   = (const float*)d_in[19];
  const float* r1a_w  = (const float*)d_in[20];
  const float* r1a_b  = (const float*)d_in[21];
  const float* r1b_w  = (const float*)d_in[22];
  const float* r1b_b  = (const float*)d_in[23];
  const float* r2a_w  = (const float*)d_in[24];
  const float* r2a_b  = (const float*)d_in[25];
  const float* r2b_w  = (const float*)d_in[26];
  const float* r2b_b  = (const float*)d_in[27];
  const float* co_w   = (const float*)d_in[28];
  const float* co_b   = (const float*)d_in[29];
  float* out = (float*)d_out;

  // ---- workspace bump allocator (buffers 256B-aligned for b128 loads) ----
  char* ws = (char*)d_ws;
  size_t off = 0;
  auto alloc = [&](size_t bytes) -> void* {
    void* p = ws + off;
    off = (off + bytes + 255) & ~(size_t)255;
    return p;
  };
  float*     tyn    = (float*)alloc((size_t)Nb * Tt * 4);
  float*     padf   = (float*)alloc((size_t)Nb * Tt * 4);
  _Float16*  qa16   = (_Float16*)alloc((size_t)Rows * Hd * 2);
  _Float16*  ka16   = (_Float16*)alloc((size_t)Rows * Hd * 2);
  _Float16*  proj16 = (_Float16*)alloc((size_t)Mp * Hd * 2);        // [Mp x 64] K-major
  _Float16*  encw16 = (_Float16*)alloc((size_t)Ee * Pvd * 2);       // [128 x 192] K-major
  _Float16*  decw16 = (_Float16*)alloc((size_t)Kk * Ee * 2);        // [128 x 128] K-major
  float*     xp     = (float*)alloc((size_t)Rows * Mp * 4);         // reused q then k
  float*     rmax   = (float*)alloc((size_t)Rows * 4);
  float*     smax   = (float*)alloc((size_t)NHb * 4);
  float*     diagq  = (float*)alloc((size_t)Rows * 4);
  float*     diagk  = (float*)alloc((size_t)Rows * 4);
  _Float16*  qf16   = (_Float16*)alloc((size_t)Rows * Mp * 2);      // [T x Mp] row-major
  _Float16*  kfT16  = (_Float16*)alloc((size_t)Rows * Mp * 2);      // [Mp x T] K-major
  _Float16*  vaT16  = (_Float16*)alloc((size_t)Rows * 2 * Hd * 2);  // [128 x T] K-major
  _Float16*  vT16   = (_Float16*)alloc((size_t)Rows * Hd * 2);      // [64 x T] K-major
  float*     kvTf   = (float*)alloc((size_t)NHb * 2 * Hd * Mp * 4); // [128 x Mp]
  _Float16*  kvT16  = (_Float16*)alloc((size_t)NHb * 2 * Hd * Mp * 2);
  float*     kfsum  = (float*)alloc((size_t)NHb * Mp * 4);
  float*     zbuf   = (float*)alloc((size_t)Rows * 4);
  float*     pclf   = (float*)alloc((size_t)Rows * 2 * Hd * 4);
  _Float16*  pv16   = (_Float16*)alloc((size_t)Rows * Pvd * 2);
  float*     encf   = (float*)alloc((size_t)Rows * Ee * 4);
  _Float16*  t16    = (_Float16*)alloc((size_t)Rows * Ee * 2);
  float*     scoreA = (float*)alloc((size_t)Rows * Kk * 4);
  float*     x1     = (float*)alloc((size_t)Nb * 48 * 512 * 64 * 4);
  float*     x2     = (float*)alloc((size_t)Nb * 48 * 512 * 64 * 4);
  float*     x3     = (float*)alloc((size_t)Nb * 48 * 512 * 64 * 4);
  float*     scoreB = (float*)alloc((size_t)Rows * Kk * 4);

  const int TPB = 256;

  // 1) mask scan + band-identity vaT + masked vT
  scan_mask_kernel<<<Nb, Tt, 0, stream>>>(mask, tyn, padf);
  build_va_kernel<<<cdiv(Rows * 128, TPB), TPB, 0, stream>>>(tyn, padf, v_att, v, vaT16, vT16);

  // 2) FAVOR inputs: xn = x * 64^-0.25; weights to f16 (already K-major)
  cvt_scale16_kernel<<<cdiv(Rows * Hd, TPB), TPB, 0, stream>>>(q_att, qa16, 0.35355339f, Rows * Hd);
  cvt_scale16_kernel<<<cdiv(Rows * Hd, TPB), TPB, 0, stream>>>(k_att, ka16, 0.35355339f, Rows * Hd);
  cvt_pad_kernel<<<cdiv(Mp * Hd, TPB), TPB, 0, stream>>>(proj, proj16, Mr, Hd, Mp);
  cvt16_kernel<<<cdiv(Ee * Pvd, TPB), TPB, 0, stream>>>(enc_w, encw16, Ee * Pvd);
  cvt16_kernel<<<cdiv(Kk * Ee, TPB), TPB, 0, stream>>>(dec_w, decw16, Kk * Ee);
  diag_kernel<<<cdiv(Rows, TPB), TPB, 0, stream>>>(q_att, diagq);
  diag_kernel<<<cdiv(Rows, TPB), TPB, 0, stream>>>(k_att, diagk);

  // 3) qf = favor(q): xp = xn @ proj^T  [1024x64]*[64x288] per head
  launch_gemm(stream, qa16, proj16, xp, Tt, Mp, Hd,
              (long)Tt * Hd, 0, (long)Tt * Mp, Hd, Hd, Mp, NHb);
  rowmax_kernel<<<Rows, 128, 0, stream>>>(xp, rmax);
  favor_exp_kernel<<<cdiv(Rows * Mp, TPB), TPB, 0, stream>>>(xp, diagq, rmax, smax, 0, 0, qf16);

  // 4) kf = favor(k): stab = max over (T, M) per head; store transposed (kfT)
  launch_gemm(stream, ka16, proj16, xp, Tt, Mp, Hd,
              (long)Tt * Hd, 0, (long)Tt * Mp, Hd, Hd, Mp, NHb);
  rowmax_kernel<<<Rows, 128, 0, stream>>>(xp, rmax);
  nhmax_kernel<<<NHb, 256, 0, stream>>>(rmax, smax);
  favor_exp_kernel<<<cdiv(Rows * Mp, TPB), TPB, 0, stream>>>(xp, diagk, rmax, smax, 1, 1, kfT16);

  // 5) kvT = va^T @ kf  ([128 x 1024]*[1024 x 288]) ; z = 1/(qf . sum_t kf)
  kfsum_kernel<<<cdiv(NHb * Mp, TPB), TPB, 0, stream>>>(kfT16, kfsum);
  zdot_kernel<<<cdiv(Rows, TPB), TPB, 0, stream>>>(qf16, kfsum, zbuf);
  launch_gemm(stream, vaT16, kfT16, kvTf, 2 * Hd, Mp, Tt,
              (long)(2 * Hd) * Tt, (long)Mp * Tt, (long)(2 * Hd) * Mp,
              Tt, Tt, Mp, NHb);
  cvt16_kernel<<<cdiv(NHb * 2 * Hd * Mp, TPB), TPB, 0, stream>>>(kvTf, kvT16, NHb * 2 * Hd * Mp);

  // 6) pcl = qf @ kv  ([1024 x 288] * [288 x 128], B = kvT K-major); pv = [pcl*z || v]
  launch_gemm(stream, qf16, kvT16, pclf, Tt, 2 * Hd, Mp,
              (long)Tt * Mp, (long)(2 * Hd) * Mp, (long)Tt * 2 * Hd,
              Mp, Mp, 2 * Hd, NHb);
  build_pv_kernel<<<cdiv(Rows * Pvd, TPB), TPB, 0, stream>>>(pclf, zbuf, vT16, pv16);

  // 7) enc GEMM (K=192) -> LN+GELU -> dec GEMM (K=128) + bias
  launch_gemm(stream, pv16, encw16, encf, Tt, Ee, Pvd,
              (long)Tt * Pvd, 0, (long)Tt * Ee, Pvd, Pvd, Ee, NHb);
  ln_gelu_kernel<<<Rows, 128, 0, stream>>>(encf, enc_b, ln_g, ln_b, t16);
  launch_gemm(stream, t16, decw16, scoreA, Tt, Kk, Ee,
              (long)Tt * Ee, 0, (long)Tt * Kk, Ee, Ee, Kk, NHb);
  add_rowbias_kernel<<<cdiv(Rows * Kk, TPB), TPB, 0, stream>>>(scoreA, dec_b, Kk, Rows * Kk);

  // 8) CNN attention predictor
  int c1_total = Nb * 48 * 512 * 64;
  conv3x3_kernel<<<cdiv(c1_total, TPB), TPB, 0, stream>>>(
      scoreA, c1_w, c1_b, nullptr, x1, Nb, 12, Tt, Kk, 48, 512, 64, 2, 1);
  conv3x3_kernel<<<cdiv(c1_total, TPB), TPB, 0, stream>>>(
      x1, r1a_w, r1a_b, nullptr, x2, Nb, 48, 512, 64, 48, 512, 64, 1, 1);
  conv3x3_kernel<<<cdiv(c1_total, TPB), TPB, 0, stream>>>(
      x2, r1b_w, r1b_b, x1, x3, Nb, 48, 512, 64, 48, 512, 64, 1, 2);
  conv3x3_kernel<<<cdiv(c1_total, TPB), TPB, 0, stream>>>(
      x3, r2a_w, r2a_b, nullptr, x2, Nb, 48, 512, 64, 48, 512, 64, 1, 1);
  conv3x3_kernel<<<cdiv(c1_total, TPB), TPB, 0, stream>>>(
      x2, r2b_w, r2b_b, x3, x1, Nb, 48, 512, 64, 48, 512, 64, 1, 2);
  int co_total = Nb * 12 * Tt * Kk;
  conv3x3_up_kernel<<<cdiv(co_total, TPB), TPB, 0, stream>>>(
      x1, co_w, co_b, scoreB, Nb, 48, 512, 64, 12);

  // 9) fused resize + mask + softmax + (P @ V) with WMMA, transposed store
  attn_out_kernel<<<dim3(Tt / 16, Hn, Nb), 128, 0, stream>>>(scoreB, mask, vT16, out);
}
